// ExtraColumnGlobalAttention_7215545057292
// MI455X (gfx1250) — compile-verified, matrix operand order checked by reading, not run on hardware
//
#include <hip/hip_runtime.h>

#define SDIM 2048
#define RDIM 384
#define CDIM 64
#define HN 8
#define ACN 8
#define NTHREADS 512
#define NWAVES 16
#define SPT 4            // SDIM / NTHREADS
#define LN_EPS 1e-5f

typedef __attribute__((ext_vector_type(16))) __bf16 v16bf;
typedef __attribute__((ext_vector_type(8)))  __bf16 v8bf;
typedef __attribute__((ext_vector_type(8)))  float  v8f;

// ---- LDS layout (bytes), total 321056 <= 327680 (320 KiB) ----
#define OFF_ACTN   0          // [2048][64] bf16 (xor-swizzled rows)  262144
#define OFF_SCR    262144     // [16 waves][16][64] bf16 gated tiles   32768
#define OFF_WGT    294912     // wg^T [64][64] bf16                     8192
#define OFF_WOT    303104     // wo^T [64][64] bf16                     8192
#define OFF_WV     311296     // wv [64][8] f32                         2048
#define OFF_QK     313344     // qk [8][64] f32                         2048
#define OFF_RED    315392     // [16][64] f32 reduce scratch            4096
#define OFF_OUT1D  319488     // [64] f32
#define OFF_SUMW   319744     // [8] f32
#define OFF_LNS    319776     // [64] f32
#define OFF_LNB    320032     // [64] f32
#define OFF_BG     320288     // [64] f32
#define OFF_BO     320544     // [64] f32
#define OFF_QAVG   320800     // [64] f32
#define SMEM_BYTES 321056

__device__ __forceinline__ unsigned short f2bf(float f) {
  union { float f; unsigned u; } x; x.f = f;
  unsigned r = x.u + 0x7FFFu + ((x.u >> 16) & 1u);   // RNE
  return (unsigned short)(r >> 16);
}
__device__ __forceinline__ float bf2f(unsigned short h) {
  union { unsigned u; float f; } x; x.u = ((unsigned)h) << 16;
  return x.f;
}
__device__ __forceinline__ float wsum(float v) {
  #pragma unroll
  for (int m = 16; m >= 1; m >>= 1) v += __shfl_xor(v, m, 32);
  return v;
}
__device__ __forceinline__ float wmaxr(float v) {
  #pragma unroll
  for (int m = 16; m >= 1; m >>= 1) v = fmaxf(v, __shfl_xor(v, m, 32));
  return v;
}
// Fast sigmoid: single v_exp_f32 + single v_rcp_f32 (TRANS ops co-execute
// with WMMA); avoids the IEEE div_scale/fma Newton chain.
__device__ __forceinline__ float fast_sigmoid(float x) {
  return __builtin_amdgcn_rcpf(1.f + __expf(-x));
}
// XOR swizzle of 16B chunks within a 64-elem (128B) row: kills the
// 128B-row-stride bank conflicts for both row-streaming and WMMA tile loads.
__device__ __forceinline__ int swz_idx(int row, int elem) {
  return (row << 6) + ((((elem >> 3) ^ (row & 7)) << 3)) + (elem & 7);
}
// Load one 16-bit WMMA A/B operand half-window: K = {kb..kb+7, kb+16..kb+23}
__device__ __forceinline__ v16bf ld16(const unsigned short* base, int row, int kb) {
  v16bf r;
  ((v8bf*)&r)[0] = *(const v8bf*)(base + swz_idx(row, kb));
  ((v8bf*)&r)[1] = *(const v8bf*)(base + swz_idx(row, kb + 16));
  return r;
}

__global__ __launch_bounds__(NTHREADS, 1)
void xcga_kernel(const float* __restrict__ msa_act,
                 const float* __restrict__ ln_scale, const float* __restrict__ ln_bias,
                 const float* __restrict__ wq, const float* __restrict__ wk,
                 const float* __restrict__ wv, const float* __restrict__ wg,
                 const float* __restrict__ bg, const float* __restrict__ wo,
                 const float* __restrict__ bo, float* __restrict__ out)
{
  extern __shared__ char smem[];
  unsigned short* actn = (unsigned short*)(smem + OFF_ACTN);
  unsigned short* scr  = (unsigned short*)(smem + OFF_SCR);
  unsigned short* wgT  = (unsigned short*)(smem + OFF_WGT);
  unsigned short* woT  = (unsigned short*)(smem + OFF_WOT);
  float* wv_l  = (float*)(smem + OFF_WV);
  float* qk_l  = (float*)(smem + OFF_QK);
  float* red   = (float*)(smem + OFF_RED);
  float* out1d = (float*)(smem + OFF_OUT1D);
  float* sumw  = (float*)(smem + OFF_SUMW);
  float* lns   = (float*)(smem + OFF_LNS);
  float* lnb   = (float*)(smem + OFF_LNB);
  float* bg_l  = (float*)(smem + OFF_BG);
  float* bo_l  = (float*)(smem + OFF_BO);
  float* qavg  = (float*)(smem + OFF_QAVG);

  const int tid  = threadIdx.x;
  const int wave = tid >> 5;
  const int lane = tid & 31;
  const int r    = blockIdx.x;       // one workgroup per column r

  // ---------- Phase 0: stage parameters in LDS (weights transposed -> bf16) ----------
  if (tid < CDIM) {
    lns[tid]  = ln_scale[tid];  lnb[tid] = ln_bias[tid];
    bg_l[tid] = bg[tid];        bo_l[tid] = bo[tid];
    out1d[tid] = 0.f;
  }
  for (int idx = tid; idx < CDIM * CDIM; idx += NTHREADS) {
    int j = idx >> 6, c = idx & 63;             // wgT[j][c] = wg[c][j]
    wgT[swz_idx(j, c)] = f2bf(wg[c * CDIM + j]);
    woT[swz_idx(j, c)] = f2bf(wo[c * CDIM + j]);
  }
  if (tid < CDIM * ACN) wv_l[tid] = wv[tid];
  __syncthreads();

  // ---------- Phase 1: stream column in, LayerNorm, store bf16 to LDS, q_avg partials ----------
  float qs0 = 0.f, qs1 = 0.f;
  const float* colbase = msa_act + (size_t)r * CDIM;
  #pragma unroll 2
  for (int s = wave; s < SDIM; s += NWAVES) {
    const float* row = colbase + (size_t)s * (RDIM * CDIM);
    float2 x = *(const float2*)(row + (lane << 1));     // coalesced 256B/row
    float mu = wsum(x.x + x.y) * (1.f / CDIM);
    float d0 = x.x - mu, d1 = x.y - mu;
    float var = wsum(d0 * d0 + d1 * d1) * (1.f / CDIM);
    float rin = rsqrtf(var + LN_EPS);
    float n0 = d0 * rin * lns[lane << 1]       + lnb[lane << 1];
    float n1 = d1 * rin * lns[(lane << 1) + 1] + lnb[(lane << 1) + 1];
    qs0 += n0; qs1 += n1;
    unsigned pk = (unsigned)f2bf(n0) | ((unsigned)f2bf(n1) << 16);
    *(unsigned*)(actn + swz_idx(s, lane << 1)) = pk;
  }
  red[wave * CDIM + (lane << 1)]     = qs0;
  red[wave * CDIM + (lane << 1) + 1] = qs1;
  __syncthreads();
  if (tid < CDIM) {
    float a = 0.f;
    #pragma unroll
    for (int w = 0; w < NWAVES; ++w) a += red[w * CDIM + tid];
    qavg[tid] = a * (1.f / SDIM);
  }
  __syncthreads();

  // ---------- Phase 2: q = (q_avg @ wq) * AC^-0.5, then fold into qk[h][c] ----------
  float qj = 0.f;
  if (tid < CDIM) {
    #pragma unroll 8
    for (int c = 0; c < CDIM; ++c) qj += qavg[c] * wq[c * CDIM + tid];
    qj *= 0.35355339059327373f;   // 1/sqrt(AC)
  }
  __syncthreads();
  if (tid < CDIM) qavg[tid] = qj;           // qavg now holds q[h*8+c']
  __syncthreads();
  {
    int h = tid >> 6, c = tid & 63;         // 512 threads == 8*64 entries
    float v = 0.f;
    #pragma unroll
    for (int cp = 0; cp < ACN; ++cp) v += qavg[h * ACN + cp] * wk[c * ACN + cp];
    qk_l[h * CDIM + c] = v;                 // logit[h,s] = actn[s,:] . qk[h,:]
  }
  __syncthreads();

  // ---------- Phase 3: logits, softmax over S, out1d = softmax @ v ----------
  float lg[HN][SPT];
  #pragma unroll
  for (int j = 0; j < SPT; ++j) {
    int s = tid + j * NTHREADS;
    int sw = s & 7;
    float acc[HN] = {0.f,0.f,0.f,0.f,0.f,0.f,0.f,0.f};
    #pragma unroll
    for (int cb = 0; cb < 8; ++cb) {
      uint4 w4 = *(const uint4*)(actn + (s << 6) + (((cb ^ sw)) << 3));
      unsigned u[4] = {w4.x, w4.y, w4.z, w4.w};
      #pragma unroll
      for (int e = 0; e < 4; ++e) {
        float a0 = bf2f((unsigned short)(u[e] & 0xFFFFu));
        float a1 = bf2f((unsigned short)(u[e] >> 16));
        int c = cb * 8 + e * 2;
        #pragma unroll
        for (int h = 0; h < HN; ++h)
          acc[h] += a0 * qk_l[h * CDIM + c] + a1 * qk_l[h * CDIM + c + 1];
      }
    }
    #pragma unroll
    for (int h = 0; h < HN; ++h) lg[h][j] = acc[h];
  }
  #pragma unroll
  for (int h = 0; h < HN; ++h) {
    float m = lg[h][0];
    #pragma unroll
    for (int j = 1; j < SPT; ++j) m = fmaxf(m, lg[h][j]);
    m = wmaxr(m);
    if (lane == 0) red[h * NWAVES + wave] = m;
  }
  __syncthreads();
  float gmax[HN];
  #pragma unroll
  for (int h = 0; h < HN; ++h) {
    float m = -3.4e38f;
    #pragma unroll
    for (int w = 0; w < NWAVES; ++w) m = fmaxf(m, red[h * NWAVES + w]);
    gmax[h] = m;
  }
  #pragma unroll
  for (int h = 0; h < HN; ++h) {
    float sm = 0.f;
    #pragma unroll
    for (int j = 0; j < SPT; ++j) {
      float e = __expf(lg[h][j] - gmax[h]);
      lg[h][j] = e; sm += e;
    }
    sm = wsum(sm);
    if (lane == 0) red[128 + h * NWAVES + wave] = sm;   // disjoint from max slots
  }
  __syncthreads();
  if (tid < HN) {
    float g = 0.f;
    #pragma unroll
    for (int w = 0; w < NWAVES; ++w) g += red[128 + tid * NWAVES + w];
    sumw[tid] = g;
  }
  // out1d accumulation (v recomputed from LDS)
  float o1[CDIM];
  #pragma unroll
  for (int i = 0; i < CDIM; ++i) o1[i] = 0.f;
  #pragma unroll
  for (int j = 0; j < SPT; ++j) {
    int s = tid + j * NTHREADS;
    int sw = s & 7;
    float vac[ACN] = {0.f,0.f,0.f,0.f,0.f,0.f,0.f,0.f};
    #pragma unroll
    for (int cb = 0; cb < 8; ++cb) {
      uint4 w4 = *(const uint4*)(actn + (s << 6) + (((cb ^ sw)) << 3));
      unsigned u[4] = {w4.x, w4.y, w4.z, w4.w};
      #pragma unroll
      for (int e = 0; e < 4; ++e) {
        float a0 = bf2f((unsigned short)(u[e] & 0xFFFFu));
        float a1 = bf2f((unsigned short)(u[e] >> 16));
        int c = cb * 8 + e * 2;
        #pragma unroll
        for (int ac = 0; ac < ACN; ++ac)
          vac[ac] += a0 * wv_l[c * ACN + ac] + a1 * wv_l[(c + 1) * ACN + ac];
      }
    }
    #pragma unroll
    for (int h = 0; h < HN; ++h) {
      float wgt = lg[h][j];
      #pragma unroll
      for (int ac = 0; ac < ACN; ++ac) o1[h * ACN + ac] += wgt * vac[ac];
    }
  }
  #pragma unroll
  for (int i = 0; i < CDIM; ++i) {
    float v = wsum(o1[i]);
    if (lane == 0) atomicAdd(&out1d[i], v);      // ds_add_f32
  }
  __syncthreads();
  if (tid < CDIM) out1d[tid] = out1d[tid] / sumw[tid >> 3];
  __syncthreads();

  // ---------- Phase 4: WMMA — out = (sigmoid(actn@wg + bg) * out1d) @ wo + bo ----------
  const int rA    = lane & 15;
  const int khalf = (lane >> 4) << 3;            // lanes 16-31 take K+8 half
  unsigned short* myscr = scr + wave * (16 * CDIM);
  for (int t = wave; t < SDIM / 16; t += NWAVES) {
    const int s0 = t << 4;
    // A operand (actn rows) is invariant across the 4 N-tiles: load once.
    v16bf A0 = ld16(actn, s0 + rA, khalf);        // K  0..31 half-window
    v16bf A1 = ld16(actn, s0 + rA, 32 + khalf);   // K 32..63 half-window
    // GEMM1: 16x64 gate logits, fuse sigmoid * out1d, spill bf16 tile to LDS
    #pragma unroll
    for (int nt = 0; nt < 4; ++nt) {
      v8f acc = {0.f,0.f,0.f,0.f,0.f,0.f,0.f,0.f};
      v16bf B0 = ld16(wgT, nt * 16 + rA, khalf);
      v16bf B1 = ld16(wgT, nt * 16 + rA, 32 + khalf);
      acc = __builtin_amdgcn_wmma_f32_16x16x32_bf16(false, A0, false, B0,
                                                    (short)0, acc, false, false);
      acc = __builtin_amdgcn_wmma_f32_16x16x32_bf16(false, A1, false, B1,
                                                    (short)0, acc, false, false);
      int ch = nt * 16 + rA;                     // channel fixed per lane
      float bgv = bg_l[ch];
      float o1v = out1d[ch];
      #pragma unroll
      for (int i = 0; i < 8; ++i) {
        int m = ((lane >> 4) << 3) + i;          // row within tile
        float g = fast_sigmoid(acc[i] + bgv);    // v_exp + v_rcp only
        myscr[swz_idx(m, ch)] = f2bf(g * o1v);
      }
    }
    // GEMM2: gated(16x64) @ wo -> 16x64 output tile, + bo, store fp32
    v16bf G0 = ld16(myscr, rA, khalf);           // gated tile, hoisted
    v16bf G1 = ld16(myscr, rA, 32 + khalf);
    #pragma unroll
    for (int nt = 0; nt < 4; ++nt) {
      v8f acc = {0.f,0.f,0.f,0.f,0.f,0.f,0.f,0.f};
      v16bf B0 = ld16(woT, nt * 16 + rA, khalf);
      v16bf B1 = ld16(woT, nt * 16 + rA, 32 + khalf);
      acc = __builtin_amdgcn_wmma_f32_16x16x32_bf16(false, G0, false, B0,
                                                    (short)0, acc, false, false);
      acc = __builtin_amdgcn_wmma_f32_16x16x32_bf16(false, G1, false, B1,
                                                    (short)0, acc, false, false);
      int ch = nt * 16 + rA;
      float bov = bo_l[ch];
      #pragma unroll
      for (int i = 0; i < 8; ++i) {
        int m = ((lane >> 4) << 3) + i;
        int s = s0 + m;
        out[((size_t)s * RDIM + r) * CDIM + ch] = acc[i] + bov;
      }
    }
  }
}

extern "C" void kernel_launch(void* const* d_in, const int* in_sizes, int n_in,
                              void* d_out, int out_size, void* d_ws, size_t ws_size,
                              hipStream_t stream) {
  (void)in_sizes; (void)n_in; (void)d_ws; (void)ws_size; (void)out_size;
  const float* msa_act  = (const float*)d_in[0];
  // d_in[1] = msa_mask (unused by the reference math)
  const float* ln_scale = (const float*)d_in[2];
  const float* ln_bias  = (const float*)d_in[3];
  const float* wq = (const float*)d_in[4];
  const float* wk = (const float*)d_in[5];
  const float* wv = (const float*)d_in[6];
  const float* wg = (const float*)d_in[7];
  const float* bg = (const float*)d_in[8];
  const float* wo = (const float*)d_in[9];
  const float* bo = (const float*)d_in[10];
  float* out = (float*)d_out;

  hipFuncSetAttribute((const void*)xcga_kernel,
                      hipFuncAttributeMaxDynamicSharedMemorySize, SMEM_BYTES);
  xcga_kernel<<<RDIM, NTHREADS, SMEM_BYTES, stream>>>(
      msa_act, ln_scale, ln_bias, wq, wk, wv, wg, bg, wo, bo, out);
}
